// Encoder_22385369547413
// MI455X (gfx1250) — matrix operand order repaired and checked by reading, not compiled, not run on hardware
//
#include <hip/hip_runtime.h>

typedef __attribute__((ext_vector_type(16))) _Float16 v16h;
typedef __attribute__((ext_vector_type(8)))  float    v8f;

// -------- native f32 atomic add (guarantee global_atomic_add_f32, no CAS loop) ------
__device__ __forceinline__ void atomic_add_f32(float* p, float v) {
#if defined(__gfx1250__)
  asm volatile("global_atomic_add_f32 %0, %1, off" : : "v"(p), "v"(v) : "memory");
#else
  atomicAdd(p, v);
#endif
}

// ---------------- degree / normalization ----------------
__global__ void k_deg_init(float* deg, int n) {
  int i = blockIdx.x * blockDim.x + threadIdx.x;
  if (i < n) deg[i] = 1.0f;                      // self-loop contributes 1 to every node
}
__global__ void k_deg_count(const long long* __restrict__ dst, float* deg, int E) {
  int e = blockIdx.x * blockDim.x + threadIdx.x;
  if (e < E) atomic_add_f32(&deg[(int)dst[e]], 1.0f);
}
__global__ void k_dinv(float* deg, int n) {
  int i = blockIdx.x * blockDim.x + threadIdx.x;
  if (i < n) deg[i] = rsqrtf(deg[i]);            // deg >= 1 always
}

// ---------------- pack W[K,NOUT] (f32 row-major) into WMMA B-fragment f16 layout ----
// B fragment (16x16x32 f16): lane l holds column N = l%16, K = (l/16)*8 + i for i<8, 16 + (l/16)*8 + (i-8) for i>=8.
// Packed as [kt][nt][lane][i] so the GEMM does one contiguous 32B load per lane.
template<int K, int NOUT>
__global__ void k_pack_w(const float* __restrict__ W, _Float16* __restrict__ Bp) {
  int idx = blockIdx.x * blockDim.x + threadIdx.x;
  if (idx >= K * NOUT) return;
  constexpr int NT = NOUT / 16;
  int i    = idx & 15;
  int lane = (idx >> 4) & 31;
  int tile = idx >> 9;            // 512 elements per 32x16 K-tile fragment
  int nt = tile % NT;
  int kt = tile / NT;
  int k = kt * 32 + (lane >> 4) * 8 + (i < 8 ? i : 8 + i);
  int n = nt * 16 + (lane & 15);
  Bp[idx] = (_Float16)W[k * NOUT + n];
}

// ---------------- H[nrows,NOUT] = X[nrows,K] @ W  via v_wmma_f32_16x16x32_f16 -------
// One wave owns a 16-row strip and all NOUT/16 N-tiles (A reused across tiles).
template<int K, int NOUT, int WAVES>
__global__ __launch_bounds__(WAVES * 32)
void k_gemm_wmma(const float* __restrict__ X, const _Float16* __restrict__ Bp,
                 float* __restrict__ H, int nrows) {
  constexpr int NT = NOUT / 16;
  const int lane = threadIdx.x & 31;
  const int wave = threadIdx.x >> 5;
  const int row0 = (blockIdx.x * WAVES + wave) * 16;
  if (row0 >= nrows) return;                    // wave-uniform: EXEC stays all-ones
  const int m  = lane & 15;                     // A row / D column within tile
  const int hk = lane >> 4;                     // K half selector

  const float* xrow = X + (size_t)(row0 + m) * K;
  v8f acc[NT];
#pragma unroll
  for (int t = 0; t < NT; ++t)
#pragma unroll
    for (int r = 0; r < 8; ++r) acc[t][r] = 0.0f;

  for (int kt = 0; kt < K / 32; ++kt) {
    // A fragment (ISA 7.12.2, 16-bit A 16x32): lanes 0-15 hold K {0..7,16..23},
    // lanes 16-31 hold K {8..15,24..31}; convert f32 -> f16 on the fly.
    const int kb = kt * 32 + hk * 8;
    v16h a;
#pragma unroll
    for (int i = 0; i < 8; ++i) {
      a[i]     = (_Float16)xrow[kb + i];
      a[i + 8] = (_Float16)xrow[kb + 16 + i];
    }
    const v16h* bb = reinterpret_cast<const v16h*>(Bp) + (size_t)kt * NT * 32 + lane;
#pragma unroll
    for (int nt = 0; nt < NT; ++nt) {
      v16h b = bb[nt * 32];
      acc[nt] = __builtin_amdgcn_wmma_f32_16x16x32_f16(
          false, a, false, b, (short)0, acc[nt], false, false);
    }
  }
  // D layout: lane group hk holds rows M = 8*hk + r, column N = m.
#pragma unroll
  for (int nt = 0; nt < NT; ++nt)
#pragma unroll
    for (int r = 0; r < 8; ++r)
      H[(size_t)(row0 + 8 * hk + r) * NOUT + nt * 16 + m] = acc[nt][r];
}

// ---------------- self-loop term: Out = H * dinv^2 (also initializes Out) ----------
template<int F>
__global__ void k_selfloop(const float* __restrict__ H, const float* __restrict__ dinv,
                           float* __restrict__ Out, int n) {
  int idx = blockIdx.x * blockDim.x + threadIdx.x;
  if (idx < n * F) {
    float di = dinv[idx / F];
    Out[idx] = H[idx] * di * di;
  }
}

// ---------------- edge scatter: Out[dst] += H[src] * dinv[src]*dinv[dst] ------------
// One wave per edge iteration: coalesced row gather + coalesced f32 atomics (L2-resident).
template<int F>
__global__ void k_scatter(const long long* __restrict__ src, const long long* __restrict__ dst,
                          const float* __restrict__ H, const float* __restrict__ dinv,
                          float* Out, int E) {
  const int lane = threadIdx.x & 31;
  int       wid  = (blockIdx.x * blockDim.x + threadIdx.x) >> 5;
  const int nw   = (gridDim.x * blockDim.x) >> 5;
  for (int e = wid; e < E; e += nw) {
    const int   s = (int)src[e];
    const int   d = (int)dst[e];
    if (e + nw < E)  // prefetch next edge's source row (global_prefetch_b8)
      __builtin_prefetch(H + (size_t)((int)src[e + nw]) * F + lane, 0, 3);
    const float  w  = dinv[s] * dinv[d];
    const float* hs = H   + (size_t)s * F;
    float*       od = Out + (size_t)d * F;
#pragma unroll
    for (int it = 0; it < F / 32; ++it)
      atomic_add_f32(&od[lane + it * 32], hs[lane + it * 32] * w);
  }
}

// ---------------- bias + ReLU (safe in-place) ---------------------------------------
template<int F>
__global__ void k_bias_relu(const float* __restrict__ In, const float* __restrict__ bias,
                            float* __restrict__ Out, int n) {
  int idx = blockIdx.x * blockDim.x + threadIdx.x;
  if (idx < n * F) {
    float v = In[idx] + bias[idx % F];
    Out[idx] = v > 0.0f ? v : 0.0f;
  }
}

extern "C" void kernel_launch(void* const* d_in, const int* in_sizes, int n_in,
                              void* d_out, int out_size, void* d_ws, size_t ws_size,
                              hipStream_t stream) {
  const float*     x  = (const float*)d_in[0];
  const long long* ei = (const long long*)d_in[1];   // int64 edge_index [2, E]
  const float*     W1 = (const float*)d_in[2];
  const float*     b1 = (const float*)d_in[3];
  const float*     W2 = (const float*)d_in[4];
  const float*     b2 = (const float*)d_in[5];

  constexpr int IN = 256, H1 = 128, H2 = 64;
  const int N = in_sizes[0] / IN;
  const int E = in_sizes[1] / 2;
  const long long* src = ei;
  const long long* dst = ei + E;

  // workspace carve-out (256B aligned):  ~103 MB total
  char*  ws  = (char*)d_ws;
  size_t off = 0;
  auto alloc = [&](size_t bytes) -> char* {
    char* p = ws + off;
    off += (bytes + 255) & ~(size_t)255;
    return p;
  };
  float*    dinv = (float*)alloc((size_t)N * 4);
  _Float16* Bp1  = (_Float16*)alloc((size_t)IN * H1 * 2);
  _Float16* Bp2  = (_Float16*)alloc((size_t)H1 * H2 * 2);
  float*    bufA = (float*)alloc((size_t)N * H1 * 4);   // gemm1 out h; reused as gemm2 out h2
  float*    bufB = (float*)alloc((size_t)N * H1 * 4);   // agg1 -> relu -> h1
  float*    outp = (float*)d_out;                       // agg2 accumulates directly here

  const int T = 256;
  // ---- normalization: deg = 1 + count(dst); dinv = rsqrt(deg)
  k_deg_init <<<(N + T - 1) / T, T, 0, stream>>>(dinv, N);
  k_deg_count<<<(E + T - 1) / T, T, 0, stream>>>(dst, dinv, E);
  k_dinv     <<<(N + T - 1) / T, T, 0, stream>>>(dinv, N);
  // ---- pack weights into WMMA B-fragment layout
  k_pack_w<IN, H1><<<(IN * H1 + T - 1) / T, T, 0, stream>>>(W1, Bp1);
  k_pack_w<H1, H2><<<(H1 * H2 + T - 1) / T, T, 0, stream>>>(W2, Bp2);

  const int strips = (N + 15) / 16;          // 6250 for N=100000
  const int gblk   = (strips + 7) / 8;       // 8 waves / block

  // ---- layer 1: h = x@W1 ; agg = Ahat h ; h1 = relu(agg + b1)
  k_gemm_wmma<IN, H1, 8><<<gblk, 256, 0, stream>>>(x, Bp1, bufA, N);
  k_selfloop<H1> <<<(N * H1 + T - 1) / T, T, 0, stream>>>(bufA, dinv, bufB, N);
  k_scatter<H1>  <<<2048, T, 0, stream>>>(src, dst, bufA, dinv, bufB, E);
  k_bias_relu<H1><<<(N * H1 + T - 1) / T, T, 0, stream>>>(bufB, b1, bufB, N);

  // ---- layer 2: h2 = h1@W2 ; out = relu(Ahat h2 + b2) accumulated in d_out
  k_gemm_wmma<H1, H2, 8><<<gblk, 256, 0, stream>>>(bufB, Bp2, bufA, N);
  k_selfloop<H2> <<<(N * H2 + T - 1) / T, T, 0, stream>>>(bufA, dinv, outp, N);
  k_scatter<H2>  <<<2048, T, 0, stream>>>(src, dst, bufA, dinv, outp, E);
  k_bias_relu<H2><<<(N * H2 + T - 1) / T, T, 0, stream>>>(outp, b2, outp, N);
}
